// GRU_DOUBLE_ATT_STACK_PARALLEL_DROPOUT_57990648430732
// MI455X (gfx1250) — matrix operand order repaired and checked by reading, not compiled, not run on hardware
//
#include <hip/hip_runtime.h>

// Problem constants (match reference)
constexpr int kB = 64, kA = 196, kR = 1024, kIN = 1024, kOUT = 9487, kL = 2, kP = 4;

typedef _Float16 f16;
typedef f16  v16h __attribute__((ext_vector_type(16)));
typedef f16  v8h  __attribute__((ext_vector_type(8)));
typedef f16  v4h  __attribute__((ext_vector_type(4)));
typedef float v8f __attribute__((ext_vector_type(8)));

// Branch-free transcendentals on hardware v_exp_f32 / v_log_f32 / v_rcp_f32.
__device__ __forceinline__ float frcp(float x)  { return __builtin_amdgcn_rcpf(x); }
__device__ __forceinline__ float fexp(float x)  { return __expf(x); }
__device__ __forceinline__ float flog(float x)  { return __logf(x); }
__device__ __forceinline__ float ftanh(float x) { return 1.0f - 2.0f * frcp(__expf(2.0f * x) + 1.0f); }
__device__ __forceinline__ float fsigm(float x) { return frcp(1.0f + __expf(-x)); }

// ---------------------------------------------------------------------------
// Fragment loaders. Layout per CDNA5 ISA 7.12.2:
//  A (16x32 f16): lane holds row m=lane&15; its 16 K-values are two contiguous
//  runs of 8: K = khalf*8 + [0,8) and K = 16 + khalf*8 + [0,8)  (khalf=lane>>4)
//  B (32x16 f16): lane holds col n=lane&15; same K order as A (two runs of 8)
//  C (16x16 f32): vgpr v holds row = v + 8*khalf, col = lane&15
// All loads unconditional (rows in-range: M % 16 == 0 in every use; columns
// clamped + value-zeroed, epilogues mask the tail).
// ---------------------------------------------------------------------------
__device__ __forceinline__ v16h load_a_frag_f32(const float* __restrict__ A,
                                                long long lda, int row0, int k0) {
  int lane = threadIdx.x & 31;
  const float* p = A + (long long)(row0 + (lane & 15)) * lda + k0 + (lane >> 4) * 8;
  float4 a0 = *(const float4*)(p);
  float4 a1 = *(const float4*)(p + 4);
  float4 b0 = *(const float4*)(p + 16);
  float4 b1 = *(const float4*)(p + 20);
  v16h out;
  out[0]  = (f16)a0.x; out[1]  = (f16)a0.y; out[2]  = (f16)a0.z; out[3]  = (f16)a0.w;
  out[4]  = (f16)a1.x; out[5]  = (f16)a1.y; out[6]  = (f16)a1.z; out[7]  = (f16)a1.w;
  out[8]  = (f16)b0.x; out[9]  = (f16)b0.y; out[10] = (f16)b0.z; out[11] = (f16)b0.w;
  out[12] = (f16)b1.x; out[13] = (f16)b1.y; out[14] = (f16)b1.z; out[15] = (f16)b1.w;
  return out;
}

__device__ __forceinline__ v16h load_a_frag_f16(const f16* __restrict__ A,
                                                long long lda, int row0, int k0) {
  int lane = threadIdx.x & 31;
  const f16* p = A + (long long)(row0 + (lane & 15)) * lda + k0 + (lane >> 4) * 8;
  v8h lo = *(const v8h*)(p);
  v8h hi = *(const v8h*)(p + 16);
  v16h out;
#pragma unroll
  for (int e = 0; e < 8; ++e) { out[e] = lo[e]; out[8 + e] = hi[e]; }
  return out;
}

__device__ __forceinline__ v16h load_b_frag(const float* __restrict__ W,
                                            long long ldb, int k0, int col0, int N) {
  int lane = threadIdx.x & 31;
  int n    = col0 + (lane & 15);
  bool nok = (n < N);
  int nc   = nok ? n : (N - 1);                  // clamp address, zero value below
  const float* p = W + (long long)(k0 + (lane >> 4) * 8) * ldb + nc;
  v16h out;
#pragma unroll
  for (int e = 0; e < 16; ++e) {
    float f = p[(long long)(e < 8 ? e : e + 8) * ldb];  // unconditional load
    out[e] = (f16)(nok ? f : 0.0f);              // v_cndmask, no branch
  }
  return out;
}

// ---------------------------------------------------------------------------
// Generic batched GEMM: C[p] (+)= A[p] @ W[p] + bias[p]
// One wave per 16x16 tile, 4 waves/block. M must be a multiple of 16.
// lda == K, ldb == N, ldc == N. Bias/accumulate are template params so the
// epilogue is branch-free straight-line code.
// ---------------------------------------------------------------------------
template <bool HasBias, bool Acc>
__global__ void wmma_gemm_kernel(const float* __restrict__ A, const float* __restrict__ W,
                                 const float* __restrict__ bias, float* __restrict__ C,
                                 int M, int N, int K,
                                 long long sA, long long sW, long long sB, long long sC) {
  int p = blockIdx.z;
  A += (long long)p * sA;
  W += (long long)p * sW;
  C += (long long)p * sC;
  if (HasBias) bias += (long long)p * sB;

  int wid   = threadIdx.x >> 5;
  int mtile = blockIdx.y * 4 + wid;
  int row0  = mtile * 16;
  int col0  = blockIdx.x * 16;
  if (row0 >= M) return;

  int lane  = threadIdx.x & 31;
  int nclmp = (col0 + (lane & 15) < N) ? (col0 + (lane & 15)) : (N - 1);

  v8f acc = {};
  for (int k0 = 0; k0 < K; k0 += 32) {
    v16h a = load_a_frag_f32(A, K, row0, k0);
    v16h b = load_b_frag(W, N, k0, col0, N);
    if (k0 + 32 < K) {  // prefetch next k-tile (global_prefetch_b8)
      __builtin_prefetch(A + (long long)(row0 + (lane & 15)) * K + k0 + 32, 0, 1);
      __builtin_prefetch(W + (long long)(k0 + 32 + (lane >> 4) * 16) * N + nclmp, 0, 1);
    }
    acc = __builtin_amdgcn_wmma_f32_16x16x32_f16(false, a, false, b, (short)0, acc,
                                                 false, false);
  }

  int khalf = lane >> 4;
  int col   = col0 + (lane & 15);
  if (col < N) {
    float bv = HasBias ? bias[col] : 0.0f;
    float* Cp = C + (long long)(row0 + 8 * khalf) * N + col;
    float cold[8];
    if (Acc) {
#pragma unroll
      for (int v = 0; v < 8; ++v) cold[v] = Cp[(long long)v * N];  // loads pipeline
    }
#pragma unroll
    for (int v = 0; v < 8; ++v) {
      float val = acc[v] + bv;
      if (Acc) val += cold[v];
      Cp[(long long)v * N] = val;
    }
  }
}

// ---------------------------------------------------------------------------
// Fused attention score:
//   sc[b,a] = sum_o tanh((att@Wa)[b,a,o] + ba[o] + ah[b,a]) * Wd[o] + bd
// A operand is the pre-converted f16 copy of att. Rows = B*A = 12544 = 784*16.
// Padded columns contribute tanh(dup)*0 (Wd masked), so no branches needed.
// ---------------------------------------------------------------------------
__global__ void attend_score_kernel(const f16* __restrict__ atth, const float* __restrict__ Wa,
                                    const float* __restrict__ ba, const float* __restrict__ ah,
                                    const float* __restrict__ Wd, const float* __restrict__ bd,
                                    float* __restrict__ sc) {
  const int M = kB * kA, N = kA, K = kR;
  int wid   = threadIdx.x >> 5;
  int mtile = blockIdx.x * 4 + wid;
  int row0  = mtile * 16;
  if (row0 >= M) return;

  int lane  = threadIdx.x & 31;
  int khalf = lane >> 4;

  // per-lane ah values for the 8 rows this lane owns (reused across all ntiles)
  float ahv[8];
#pragma unroll
  for (int v = 0; v < 8; ++v) ahv[v] = ah[row0 + v + 8 * khalf];

  float ps[8];
#pragma unroll
  for (int v = 0; v < 8; ++v) ps[v] = 0.0f;

  const int ntiles = (N + 15) / 16;  // 13
  for (int nt = 0; nt < ntiles; ++nt) {
    int col0 = nt * 16;
    int col  = col0 + (lane & 15);
    bool cok = (col < N);
    int cc   = cok ? col : (N - 1);
    float wdv = Wd[cc];              // unconditional (clamped) loads, issued early
    float bc  = ba[cc];
    float wv  = cok ? wdv : 0.0f;    // mask the value, not the load

    v8f acc = {};
    for (int k0 = 0; k0 < K; k0 += 32) {
      v16h a = load_a_frag_f16(atth, K, row0, k0);
      v16h b = load_b_frag(Wa, N, k0, col0, N);
      acc = __builtin_amdgcn_wmma_f32_16x16x32_f16(false, a, false, b, (short)0, acc,
                                                   false, false);
    }
#pragma unroll
    for (int v = 0; v < 8; ++v) ps[v] += ftanh(acc[v] + bc + ahv[v]) * wv;
  }
  float bd0 = bd[0];
#pragma unroll
  for (int v = 0; v < 8; ++v) {
    float s = ps[v];
    s += __shfl_xor(s, 1, 16);
    s += __shfl_xor(s, 2, 16);
    s += __shfl_xor(s, 4, 16);
    s += __shfl_xor(s, 8, 16);
    if ((lane & 15) == 0) {
      int row = row0 + v + 8 * khalf;
      sc[row] = s + bd0;
    }
  }
}

// fp32 -> f16 bulk convert (4 elements/thread, n % 4 == 0)
__global__ void f32_to_f16_kernel(const float* __restrict__ src, f16* __restrict__ dst,
                                  long long n4) {
  long long i = (long long)blockIdx.x * blockDim.x + threadIdx.x;
  if (i >= n4) return;
  float4 v = ((const float4*)src)[i];
  v4h h;
  h[0] = (f16)v.x; h[1] = (f16)v.y; h[2] = (f16)v.z; h[3] = (f16)v.w;
  ((v4h*)dst)[i] = h;
}

// Softmax over cols (A=196) per row, one wave per row.
__global__ void softmax_rows(const float* __restrict__ sc, float* __restrict__ wgt, int cols) {
  int row  = blockIdx.x;
  int lane = threadIdx.x;
  const float* src = sc + (long long)row * cols;
  float m = -3.4e38f;
  for (int j = lane; j < cols; j += 32) m = fmaxf(m, src[j]);
#pragma unroll
  for (int o = 16; o > 0; o >>= 1) m = fmaxf(m, __shfl_xor(m, o, 32));
  float ssum = 0.0f;
  for (int j = lane; j < cols; j += 32) ssum += fexp(src[j] - m);
#pragma unroll
  for (int o = 16; o > 0; o >>= 1) ssum += __shfl_xor(ssum, o, 32);
  float inv = frcp(ssum);
  for (int j = lane; j < cols; j += 32)
    wgt[(long long)row * cols + j] = fexp(src[j] - m) * inv;
}

// att_res[b,r] = sum_a att[b,a,r] * wgt[b,a]   (rank-1 per b; VALU-friendly)
__global__ void att_apply_kernel(const float* __restrict__ att, const float* __restrict__ wgt,
                                 float* __restrict__ outv) {
  int b = blockIdx.y;
  int r = blockIdx.x * blockDim.x + threadIdx.x;
  const float* attb = att + (long long)b * kA * kR;
  const float* wb   = wgt + (long long)b * kA;
  float acc = 0.0f;
  for (int a = 0; a < kA; ++a) acc += attb[(long long)a * kR + r] * wb[a];
  outv[(long long)b * kR + r] = acc;
}

__global__ void add2_kernel(const float* __restrict__ a, const float* __restrict__ b,
                            float* __restrict__ c, int n) {
  int i = blockIdx.x * blockDim.x + threadIdx.x;
  if (i < n) c[i] = a[i] + b[i];
}

// z = sigmoid(sums[...,:R]); rgh = sigmoid(sums[...,R:2R]) * prev_h
__global__ void gate_kernel(const float* __restrict__ sums, const float* __restrict__ prev_h,
                            float* __restrict__ z, float* __restrict__ rgh) {
  int i = blockIdx.x * blockDim.x + threadIdx.x;  // over P*B*R
  if (i >= kP * kB * kR) return;
  int r  = i % kR;
  int pb = i / kR;
  int b  = pb % kB;
  long long base = (long long)pb * (2 * kR);
  float zv = fsigm(sums[base + r]);
  float rv = fsigm(sums[base + kR + r]);
  z[i]   = zv;
  rgh[i] = rv * prev_h[(long long)b * kR + r];
}

// next_h = mean_p((1-z)*prev_h + z*tanh(hgpre))
__global__ void nh_kernel(const float* __restrict__ z, const float* __restrict__ hgpre,
                          const float* __restrict__ prev_h, float* __restrict__ next_h) {
  int i = blockIdx.x * blockDim.x + threadIdx.x;  // over B*R
  if (i >= kB * kR) return;
  float ph = prev_h[i];
  float s = 0.0f;
#pragma unroll
  for (int p = 0; p < kP; ++p) {
    long long idx = (long long)p * (kB * kR) + i;
    float hg = ftanh(hgpre[idx]);
    float zv = z[idx];
    s += (1.0f - zv) * ph + zv * hg;
  }
  next_h[i] = s * (1.0f / kP);
}

__global__ void log_softmax_kernel(const float* __restrict__ logits, float* __restrict__ out,
                                   int N) {
  __shared__ float red[256];
  int b = blockIdx.x, tid = threadIdx.x;
  const float* src = logits + (long long)b * N;
  float m = -3.4e38f;
  for (int j = tid; j < N; j += 256) m = fmaxf(m, src[j]);
  red[tid] = m; __syncthreads();
  for (int s2 = 128; s2 > 0; s2 >>= 1) {
    if (tid < s2) red[tid] = fmaxf(red[tid], red[tid + s2]);
    __syncthreads();
  }
  m = red[0]; __syncthreads();
  float ssum = 0.0f;
  for (int j = tid; j < N; j += 256) ssum += fexp(src[j] - m);
  red[tid] = ssum; __syncthreads();
  for (int s2 = 128; s2 > 0; s2 >>= 1) {
    if (tid < s2) red[tid] += red[tid + s2];
    __syncthreads();
  }
  float lse = m + flog(red[0]);
  for (int j = tid; j < N; j += 256) out[(long long)b * N + j] = src[j] - lse;
}

// ---------------------------------------------------------------------------
extern "C" void kernel_launch(void* const* d_in, const int* in_sizes, int n_in,
                              void* d_out, int out_size, void* d_ws, size_t ws_size,
                              hipStream_t stream) {
  (void)in_sizes; (void)n_in; (void)out_size; (void)ws_size;
  const float* x      = (const float*)d_in[0];
  const float* att    = (const float*)d_in[1];
  const float* inputs = (const float*)d_in[2];
  const float* Wa2a  = (const float*)d_in[3];  const float* ba2a  = (const float*)d_in[4];
  const float* Wh2a  = (const float*)d_in[5];  const float* bh2a  = (const float*)d_in[6];
  const float* Wd2d  = (const float*)d_in[7];  const float* bd2d  = (const float*)d_in[8];
  const float* Wa2a1 = (const float*)d_in[9];  const float* ba2a1 = (const float*)d_in[10];
  const float* Wh2a1 = (const float*)d_in[11]; const float* bh2a1 = (const float*)d_in[12];
  const float* Wd2d1 = (const float*)d_in[13]; const float* bd2d1 = (const float*)d_in[14];
  const float* Wi2h  = (const float*)d_in[15]; const float* bi2h  = (const float*)d_in[16];
  const float* Wh2h  = (const float*)d_in[17]; const float* bh2h  = (const float*)d_in[18];
  const float* Wa2h  = (const float*)d_in[19]; const float* ba2h  = (const float*)d_in[20];
  const float* Wi2h1 = (const float*)d_in[21]; const float* bi2h1 = (const float*)d_in[22];
  const float* Wh2h1 = (const float*)d_in[23]; const float* bh2h1 = (const float*)d_in[24];
  const float* Wa2h1 = (const float*)d_in[25]; const float* ba2h1 = (const float*)d_in[26];
  const float* Wproj = (const float*)d_in[27]; const float* bproj = (const float*)d_in[28];

  float* ws = (float*)d_ws;
  float* ah_buf  = ws; ws += kB * kA;
  float* sc      = ws; ws += kB * kA;
  float* wgt     = ws; ws += kB * kA;
  float* attres  = ws; ws += kB * kR;
  float* attres2 = ws; ws += kB * kR;
  float* xt      = ws; ws += kB * kIN;
  float* sums    = ws; ws += (long long)kP * kB * 2 * kR;
  float* zbuf    = ws; ws += (long long)kP * kB * kR;
  float* rgh     = ws; ws += (long long)kP * kB * kR;
  float* hgpre   = ws; ws += (long long)kP * kB * kR;
  float* next_h  = ws; ws += kB * kR;
  float* logits  = ws; ws += (long long)kB * kOUT;
  f16*   atth    = (f16*)ws;  // f16 copy of att: B*A*R elements

  float* outbuf = (float*)d_out;  // [L,B,R] then [B,OUT]
  float* out0 = outbuf;
  float* out1 = outbuf + kB * kR;
  float* lsm  = outbuf + 2 * kB * kR;

  // One-time (per launch) downcast of att for the heavy fused-score GEMM.
  {
    long long n4 = (long long)kB * kA * kR / 4;
    f32_to_f16_kernel<<<(unsigned)((n4 + 255) / 256), 256, 0, stream>>>(att, atth, n4);
  }

  // accum=0: C = A@W + bias ; accum=1: C += A@W + bias. Bias always present here.
  auto gemm = [&](const float* Ap, const float* Wp, const float* biasp, float* Cp,
                  int M, int N, int K, long long sA, long long sW, long long sB,
                  long long sC, int batches, int accum) {
    dim3 grid((N + 15) / 16, (M + 63) / 64, batches);
    if (accum)
      wmma_gemm_kernel<true, true><<<grid, 128, 0, stream>>>(Ap, Wp, biasp, Cp, M, N, K,
                                                             sA, sW, sB, sC);
    else
      wmma_gemm_kernel<true, false><<<grid, 128, 0, stream>>>(Ap, Wp, biasp, Cp, M, N, K,
                                                              sA, sW, sB, sC);
  };

  auto attend = [&](const float* h, const float* Wa, const float* ba, const float* Wh,
                    const float* bh, const float* Wd, const float* bd, float* res) {
    // ah = h @ Wh + bh : [B, A]
    gemm(h, Wh, bh, ah_buf, kB, kA, kR, 0, 0, 0, 0, 1, 0);
    // sc = fused tanh((att@Wa)+ba+ah) . Wd + bd
    attend_score_kernel<<<dim3((kB * kA / 16 + 3) / 4), 128, 0, stream>>>(atth, Wa, ba,
                                                                          ah_buf, Wd, bd, sc);
    softmax_rows<<<kB, 32, 0, stream>>>(sc, wgt, kA);
    att_apply_kernel<<<dim3(kR / 256, kB), 256, 0, stream>>>(att, wgt, res);
  };

  for (int i = 0; i < kL; ++i) {
    const float* prev_h = inputs + (long long)(2 * i) * kB * kR;
    const float* xcur;
    if (i == 0) {
      xcur = x;
    } else {
      add2_kernel<<<(kB * kIN + 255) / 256, 256, 0, stream>>>(x, out0, xt, kB * kIN);
      xcur = xt;
    }

    attend(prev_h, Wa2a, ba2a, Wh2a, bh2a, Wd2d, bd2d, attres);

    // sums[p] = xt@Wi2h[p] + prev_h@Wh2h[p] + attres@Wa2h[p] + biases  [P,B,2R]
    gemm(xcur,   Wi2h, bi2h, sums, kB, 2 * kR, kIN, 0, (long long)kIN * 2 * kR, 2 * kR,
         (long long)kB * 2 * kR, kP, 0);
    gemm(prev_h, Wh2h, bh2h, sums, kB, 2 * kR, kR, 0, (long long)kR * 2 * kR, 2 * kR,
         (long long)kB * 2 * kR, kP, 1);
    gemm(attres, Wa2h, ba2h, sums, kB, 2 * kR, kR, 0, (long long)kR * 2 * kR, 2 * kR,
         (long long)kB * 2 * kR, kP, 1);
    gate_kernel<<<(kP * kB * kR + 255) / 256, 256, 0, stream>>>(sums, prev_h, zbuf, rgh);

    // hgpre[p] = xt@Wi2h1[p] + rgh[p]@Wh2h1[p] + attres@Wa2h1[p] + biases  [P,B,R]
    gemm(xcur,   Wi2h1, bi2h1, hgpre, kB, kR, kIN, 0, (long long)kIN * kR, kR,
         (long long)kB * kR, kP, 0);
    gemm(rgh,    Wh2h1, bh2h1, hgpre, kB, kR, kR, (long long)kB * kR, (long long)kR * kR,
         kR, (long long)kB * kR, kP, 1);
    gemm(attres, Wa2h1, ba2h1, hgpre, kB, kR, kR, 0, (long long)kR * kR, kR,
         (long long)kB * kR, kP, 1);
    nh_kernel<<<(kB * kR + 255) / 256, 256, 0, stream>>>(zbuf, hgpre, prev_h, next_h);

    attend(next_h, Wa2a1, ba2a1, Wh2a1, bh2a1, Wd2d1, bd2d1, attres2);

    float* outi = (i == 0) ? out0 : out1;
    add2_kernel<<<(kB * kR + 255) / 256, 256, 0, stream>>>(attres2, next_h, outi, kB * kR);
  }

  // logsoftmax(out1 @ Wproj + bproj)
  gemm(out1, Wproj, bproj, logits, kB, kOUT, kR, 0, 0, 0, 0, 1, 0);
  log_softmax_kernel<<<kB, 256, 0, stream>>>(logits, lsm, kOUT);
}
